// AttFP_41180146434472
// MI455X (gfx1250) — matrix operand order repaired and checked by reading, not compiled, not run on hardware
//
#include <hip/hip_runtime.h>
#include <math.h>
#include <limits>

// ---------------------------------------------------------------------------
// CDNA5 (gfx1250) AttentiveFP forward.  All dense GEMMs go through
// v_wmma_f32_16x16x32_f16 (f16 inputs staged in LDS, f32 accumulation).
// ---------------------------------------------------------------------------

typedef __attribute__((ext_vector_type(16))) _Float16 v16h;
typedef __attribute__((ext_vector_type(8)))  _Float16 v8h;
typedef __attribute__((ext_vector_type(8)))  float    v8f;

#define TN 256          // output columns per block
#define TM 32           // output rows per block
#define NEG_SLOPE 0.01f

#define CAT16(lo, hi) __builtin_shufflevector((lo), (hi), 0,1,2,3,4,5,6,7,8,9,10,11,12,13,14,15)

// activation codes
#define ACT_NONE  0
#define ACT_LRELU 1
#define ACT_RELU  2
#define ACT_SILU  3
// A-operand modes
#define AM_PLAIN 0
#define AM_ELU   2      // apply elu() to A elements on load (GRU input x = elu(h))

// ---------------------------------------------------------------------------
// Generic WMMA GEMM:  C[M x Nc] = act( A[M x K] * W[Nc x K]^T + bias )
//   - optional row gather + concat for A (edge message GEMM):
//       gidx != null:  A_row(m) = [ A[gidx[m], 0:K1] , A2[m, 0:K-K1] ]
//   - optional ELU applied to A elements (amode == AM_ELU)
//   - output written at C[m*ldc + coff + n]
// Block: 256 threads = 8 waves.  Tile = 32 rows x 256 cols.  Each wave owns
// two 16-col tiles and both 16-row tiles -> 4 WMMAs / K-chunk / wave.
// ---------------------------------------------------------------------------
__global__ __launch_bounds__(256)
void wmma_gemm_kernel(const float* __restrict__ A, int lda,
                      const float* __restrict__ A2, int lda2, int K1,
                      const int* __restrict__ gidx,
                      const float* __restrict__ W, int ldw,
                      const float* __restrict__ bias,
                      float* __restrict__ C, int ldc, int coff,
                      int M, int Nc, int K, int amode, int act)
{
    // row strides of 40 f16 = 80 bytes (multiple of 16B -> aligned b128 loads)
    __shared__ __align__(16) _Float16 sA[TM][40];       // A tile, row-major
    __shared__ __align__(16) _Float16 sB[TN][40];       // W^T tile, col-major: sB[n][k]

    const int tid  = threadIdx.x;
    const int lane = tid & 31;
    const int wave = tid >> 5;
    const int m0   = blockIdx.x * TM;
    const int n0   = blockIdx.y * TN;

    v8f acc00 = {}, acc01 = {}, acc10 = {}, acc11 = {};

    const int nK = (K + 31) >> 5;
    for (int kc = 0; kc < nK; ++kc) {
        const int k0 = kc * 32;

        // ---- stage A tile (32 x 32) as f16, float4 global loads ----
        {
            int i  = tid;                      // exactly (32*32)/4 = 256 slots
            int r  = i >> 3;
            int kq = (i & 7) * 4;
            int gm = m0 + r;
            int gk = k0 + kq;
            float4 v = make_float4(0.f, 0.f, 0.f, 0.f);
            float* pv = (float*)&v;
            if (gm < M) {
                if (!gidx) {
                    if (gk + 3 < K) v = *(const float4*)(A + (size_t)gm * lda + gk);
                    else {
                        for (int q = 0; q < 4; ++q)
                            if (gk + q < K) pv[q] = A[(size_t)gm * lda + gk + q];
                    }
                } else {
                    // concat-gather: [A[gidx[gm],0:K1], A2[gm,0:K-K1]]
                    // K-chunks are 32-aligned and K1 is a multiple of 32, so a
                    // float4 never straddles the split.
                    if (gk + 3 < K1) {
                        int row = gidx[gm];
                        v = *(const float4*)(A + (size_t)row * lda + gk);
                    } else if (gk >= K1) {
                        for (int q = 0; q < 4; ++q) {
                            int kk = gk + q;
                            if (kk < K) pv[q] = A2[(size_t)gm * lda2 + (kk - K1)];
                        }
                    } else {
                        int row = gidx[gm];
                        for (int q = 0; q < 4; ++q) {
                            int kk = gk + q;
                            if (kk < K1)      pv[q] = A[(size_t)row * lda + kk];
                            else if (kk < K)  pv[q] = A2[(size_t)gm * lda2 + (kk - K1)];
                        }
                    }
                }
                if (amode == AM_ELU) {
                    for (int q = 0; q < 4; ++q)
                        pv[q] = (pv[q] > 0.f) ? pv[q] : (__expf(pv[q]) - 1.f);
                }
            }
            _Float16* dp = &sA[r][kq];
            for (int q = 0; q < 4; ++q) dp[q] = (_Float16)pv[q];
            // prefetch next K-chunk of A (plain path)
            if (!gidx && kc + 1 < nK) {
                int pm = m0 + (tid & 31);
                if (pm < M && k0 + 32 < K)
                    __builtin_prefetch(A + (size_t)pm * lda + k0 + 32, 0, 1);
            }
        }
        // ---- stage B tile col-major: sB[n][k] = W[(n0+n)*ldw + k0+k] ----
        for (int i = tid; i < (TN * 32) / 4; i += 256) {
            int n  = i >> 3;
            int kq = (i & 7) * 4;
            int gn = n0 + n;
            int gk = k0 + kq;
            float4 v = make_float4(0.f, 0.f, 0.f, 0.f);
            float* pv = (float*)&v;
            if (gn < Nc) {
                if (gk + 3 < K) v = *(const float4*)(W + (size_t)gn * ldw + gk);
                else {
                    for (int q = 0; q < 4; ++q)
                        if (gk + q < K) pv[q] = W[(size_t)gn * ldw + gk + q];
                }
            }
            _Float16* dp = &sB[n][kq];
            for (int q = 0; q < 4; ++q) dp[q] = (_Float16)pv[q];
        }
        __syncthreads();

        // ---- fragments: all b128 LDS loads (ISA 7.12.2 16-bit layouts) ----
        const int mlo = lane & 15;
        const int kh  = (lane >> 4) * 8;     // A: hi lanes hold K+8
        const int kb  = (lane >> 4) * 8;     // B: hi lanes hold K+8
        v16h af0 = CAT16(*(const v8h*)(&sA[mlo][kh]),
                         *(const v8h*)(&sA[mlo][16 + kh]));
        v16h af1 = CAT16(*(const v8h*)(&sA[16 + mlo][kh]),
                         *(const v8h*)(&sA[16 + mlo][16 + kh]));
        {
            const int n = wave * 32 + mlo;
            v16h bf = CAT16(*(const v8h*)(&sB[n][kb]),
                            *(const v8h*)(&sB[n][kb + 16]));
            acc00 = __builtin_amdgcn_wmma_f32_16x16x32_f16(false, af0, false, bf,
                                                           (short)0, acc00, false, false);
            acc10 = __builtin_amdgcn_wmma_f32_16x16x32_f16(false, af1, false, bf,
                                                           (short)0, acc10, false, false);
        }
        {
            const int n = wave * 32 + 16 + mlo;
            v16h bf = CAT16(*(const v8h*)(&sB[n][kb]),
                            *(const v8h*)(&sB[n][kb + 16]));
            acc01 = __builtin_amdgcn_wmma_f32_16x16x32_f16(false, af0, false, bf,
                                                           (short)0, acc01, false, false);
            acc11 = __builtin_amdgcn_wmma_f32_16x16x32_f16(false, af1, false, bf,
                                                           (short)0, acc11, false, false);
        }
        __syncthreads();
    }

    // ---- epilogue: C/D layout (VGPR r -> row r (+8 for hi lanes); lane -> col)
    const int rowadd = (lane >> 4) * 8;
#pragma unroll
    for (int rt = 0; rt < 2; ++rt) {
#pragma unroll
        for (int t = 0; t < 2; ++t) {
            const int gn = n0 + wave * 32 + t * 16 + (lane & 15);
            if (gn >= Nc) continue;
            v8f acc = rt ? (t ? acc11 : acc10) : (t ? acc01 : acc00);
#pragma unroll
            for (int r = 0; r < 8; ++r) {
                const int gm = m0 + rt * 16 + r + rowadd;
                if (gm >= M) continue;
                float v = acc[r];
                if (bias) v += bias[gn];
                if (act == ACT_LRELU)      v = (v > 0.f) ? v : NEG_SLOPE * v;
                else if (act == ACT_RELU)  v = (v > 0.f) ? v : 0.f;
                else if (act == ACT_SILU)  v = v / (1.f + __expf(-v));
                C[(size_t)gm * ldc + coff + gn] = v;
            }
        }
    }
}

// ---------------------------------------------------------------------------
// elementwise / graph kernels
// ---------------------------------------------------------------------------
__global__ void k_fill(float* p, float v, int n) {
    int i = blockIdx.x * 256 + threadIdx.x;
    if (i < n) p[i] = v;
}

__global__ void k_relu(float* p, long long n) {
    long long i = (long long)blockIdx.x * 256 + threadIdx.x;
    if (i < n) { float v = p[i]; p[i] = v > 0.f ? v : 0.f; }
}

// out[r][c] = b[c]   (one block per row)
__global__ void k_bias_rows(float* out, const float* __restrict__ b, int C) {
    int r = blockIdx.x, c = threadIdx.x;
    if (c < C) out[(size_t)r * C + c] = b[c];
}

// y[row] = dot(X[row, 0:len], v) (+ addb[0] if addb)
__global__ __launch_bounds__(256)
void k_rowdot(const float* __restrict__ X, int ld, const float* __restrict__ v,
              float* __restrict__ y, int rows, int len, const float* __restrict__ addb)
{
    int row  = blockIdx.x * 8 + (threadIdx.x >> 5);
    int lane = threadIdx.x & 31;
    if (row >= rows) return;
    const float* xp = X + (size_t)row * ld;
    float s = 0.f;
    for (int k = lane; k < len; k += 32) s += xp[k] * v[k];
#pragma unroll
    for (int off = 16; off; off >>= 1) s += __shfl_down(s, off, 32);
    if (lane == 0) y[row] = s + (addb ? addb[0] : 0.f);
}

// a[i] = lrelu( v0[i] + v1[i1?i1[i]:i] + v2[i2?i2[i]:i] ), null terms skipped
__global__ void k_logit(float* a, int n, const float* v0,
                        const float* v1, const int* i1,
                        const float* v2, const int* i2)
{
    int i = blockIdx.x * 256 + threadIdx.x;
    if (i >= n) return;
    float s = 0.f;
    if (v0) s += v0[i];
    if (v1) s += v1[i1 ? i1[i] : i];
    if (v2) s += v2[i2 ? i2[i] : i];
    a[i] = s > 0.f ? s : NEG_SLOPE * s;
}

__device__ inline void atomicMaxFloat(float* addr, float val) {
    unsigned int* ua = (unsigned int*)addr;
    unsigned int old = *ua;
    while (true) {
        float f = __uint_as_float(old);
        if (f >= val) break;
        unsigned int assumed = old;
        old = atomicCAS(ua, assumed, __float_as_uint(val));
        if (old == assumed) break;
    }
}

__global__ void k_seg_max(const float* a, const int* seg, float* mx, int n) {
    int i = blockIdx.x * 256 + threadIdx.x;
    if (i < n) atomicMaxFloat(&mx[seg[i]], a[i]);
}

__global__ void k_seg_expsum(const float* a, const int* seg, const float* mx,
                             float* e, float* sm, int n) {
    int i = blockIdx.x * 256 + threadIdx.x;
    if (i >= n) return;
    float m = mx[seg[i]];
    if (!(m > -1e37f && m < 1e37f)) m = 0.f;   // jnp.where(isfinite(m), m, 0)
    float v = __expf(a[i] - m);
    e[i] = v;
    atomicAdd(&sm[seg[i]], v);
}

__global__ void k_seg_norm(float* e, const int* seg, const float* sm, int n) {
    int i = blockIdx.x * 256 + threadIdx.x;
    if (i < n) e[i] = e[i] / (sm[seg[i]] + 1e-16f);
}

// out[seg[i]][c] += T[(g?g[i]:i)][c] * (scale?scale[i]:1)   (block = row i)
__global__ void k_scatter(const float* __restrict__ T, const int* __restrict__ g,
                          const int* __restrict__ seg, const float* __restrict__ scale,
                          float* __restrict__ out, int C)
{
    int i = blockIdx.x, c = threadIdx.x;
    if (c >= C) return;
    int r = g ? g[i] : i;
    float v = T[(size_t)r * C + c];
    if (scale) v *= scale[i];
    atomicAdd(&out[(size_t)seg[i] * C + c], v);
}

// GRUCell combine + trailing ReLU (all GRU uses in the model are relu-wrapped)
__global__ void k_gru(const float* __restrict__ gi, const float* __restrict__ gh,
                      float* __restrict__ h, int H)
{
    int r = blockIdx.x, c = threadIdx.x;
    if (c >= H) return;
    size_t b3 = (size_t)r * 3 * H;
    float ir = gi[b3 + c], iz = gi[b3 + H + c], in = gi[b3 + 2 * H + c];
    float hr = gh[b3 + c], hz = gh[b3 + H + c], hn = gh[b3 + 2 * H + c];
    float rg = 1.f / (1.f + __expf(-(ir + hr)));
    float z  = 1.f / (1.f + __expf(-(iz + hz)));
    float nn = tanhf(in + rg * hn);
    float hp = h[(size_t)r * H + c];
    float v  = (1.f - z) * nn + z * hp;
    h[(size_t)r * H + c] = v > 0.f ? v : 0.f;
}

// ---------------------------------------------------------------------------
extern "C" void kernel_launch(void* const* d_in, const int* in_sizes, int n_in,
                              void* d_out, int out_size, void* d_ws, size_t ws_size,
                              hipStream_t stream)
{
    const int Fn = 64, Fe = 16, Fm = 200, H = 256;
    const int N = in_sizes[0] / Fn;
    const int E = in_sizes[1] / Fe;
    const int G = in_sizes[2] / Fm;

    const float* x          = (const float*)d_in[0];
    const float* edge_attr  = (const float*)d_in[1];
    const float* mol_feats  = (const float*)d_in[2];
    const float* lin1_w     = (const float*)d_in[3];
    const float* lin1_b     = (const float*)d_in[4];
    const float* gate_lin1w = (const float*)d_in[5];
    const float* gate_lin2w = (const float*)d_in[6];
    const float* gate_att_l = (const float*)d_in[7];
    const float* gate_att_r = (const float*)d_in[8];
    const float* gate_bias  = (const float*)d_in[9];
    const float* grus_wi    = (const float*)d_in[10];
    const float* grus_wh    = (const float*)d_in[11];
    const float* grus_bi    = (const float*)d_in[12];
    const float* grus_bh    = (const float*)d_in[13];
    const float* atom_w     = (const float*)d_in[14];
    const float* atom_asrc  = (const float*)d_in[15];
    const float* atom_adst  = (const float*)d_in[16];
    const float* atom_bias  = (const float*)d_in[17];
    const float* mol_w      = (const float*)d_in[18];
    const float* mol_asrc   = (const float*)d_in[19];
    const float* mol_adst   = (const float*)d_in[20];
    const float* mol_bias   = (const float*)d_in[21];
    const float* lin2_w     = (const float*)d_in[22];
    const float* lin2_b     = (const float*)d_in[23];
    const float* fcm_w[3]   = {(const float*)d_in[24], (const float*)d_in[26], (const float*)d_in[28]};
    const float* fcm_b[3]   = {(const float*)d_in[25], (const float*)d_in[27], (const float*)d_in[29]};
    const float* fc0_w = (const float*)d_in[30]; const float* fc0_b = (const float*)d_in[31];
    const float* fc1_w = (const float*)d_in[32]; const float* fc1_b = (const float*)d_in[33];
    const float* fc2_w = (const float*)d_in[34]; const float* fc2_b = (const float*)d_in[35];
    const float* fc3_w = (const float*)d_in[36]; const float* fc3_b = (const float*)d_in[37];
    const float* fc4_w = (const float*)d_in[38]; const float* fc4_b = (const float*)d_in[39];
    const int*   src   = (const int*)d_in[40];
    const int*   dst   = ((const int*)d_in[40]) + E;
    const int*   batch = (const int*)d_in[41];

    float* out = (float*)d_out;           // [G] float32
    float* ws  = (float*)d_ws;

    // ---- workspace layout (floats) -------------------------------------
    float* xh   = ws;                                  // N*H
    float* hagg = xh + (size_t)N * H;                  // N*H
    float* big  = hagg + (size_t)N * H;                // max(2*E*H, 2*N*768) = 2*N*768
    float* eA = big;                                   // E*H  (edge msgs / xl)
    float* eB = big + (size_t)E * H;                   // E*H
    float* gi = big;                                   // N*768 (overlays eA after use)
    float* gh = big + (size_t)N * 768;                 // N*768
    float* sc = big + 2 * (size_t)N * 768;
    float* aE   = sc;  sc += E;                        // logits (edges / nodes)
    float* wE   = sc;  sc += E;                        // softmax weights
    float* smax = sc;  sc += N;
    float* ssum = sc;  sc += N;
    float* ns1  = sc;  sc += N;                        // per-node scalar 1
    float* ns2  = sc;  sc += N;                        // per-node scalar 2
    float* outG = sc;  sc += (size_t)G * H;
    float* tmpG = sc;  sc += (size_t)G * H;
    float* giG  = sc;  sc += (size_t)G * 768;
    float* ghG  = sc;  sc += (size_t)G * 768;
    float* fused= sc;  sc += (size_t)G * 512;
    float* mA   = sc;  sc += (size_t)G * 512;
    float* mB   = sc;  sc += (size_t)G * 512;
    float* h128 = sc;  sc += (size_t)G * 128;
    float* dG   = sc;  sc += G;

    auto gemm = [&](const float* A, int lda, const float* A2, int lda2, int K1,
                    const int* gidx, const float* W, int ldw, const float* bias,
                    float* C, int ldc, int coff, int M, int Nc, int K,
                    int amode, int act) {
        dim3 g((M + TM - 1) / TM, (Nc + TN - 1) / TN);
        wmma_gemm_kernel<<<g, 256, 0, stream>>>(A, lda, A2, lda2, K1, gidx, W, ldw,
                                                bias, C, ldc, coff, M, Nc, K, amode, act);
    };
    auto segsoftmax = [&](float* a, float* w, const int* seg, int n, int nseg) {
        const float ninf = -std::numeric_limits<float>::infinity();
        k_fill<<<(nseg + 255) / 256, 256, 0, stream>>>(smax, ninf, nseg);
        k_fill<<<(nseg + 255) / 256, 256, 0, stream>>>(ssum, 0.f, nseg);
        k_seg_max<<<(n + 255) / 256, 256, 0, stream>>>(a, seg, smax, n);
        k_seg_expsum<<<(n + 255) / 256, 256, 0, stream>>>(a, seg, smax, w, ssum, n);
        k_seg_norm<<<(n + 255) / 256, 256, 0, stream>>>(w, seg, ssum, n);
    };
    auto gru = [&](const float* hin, float* hstate, int rows, int layer,
                   float* gib, float* ghb) {
        const float* wi = grus_wi + (size_t)layer * 768 * H;
        const float* wh = grus_wh + (size_t)layer * 768 * H;
        const float* bi = grus_bi + (size_t)layer * 768;
        const float* bh = grus_bh + (size_t)layer * 768;
        gemm(hin,    H, nullptr, 0, H, nullptr, wi, H, bi, gib, 768, 0, rows, 768, H, AM_ELU,   ACT_NONE);
        gemm(hstate, H, nullptr, 0, H, nullptr, wh, H, bh, ghb, 768, 0, rows, 768, H, AM_PLAIN, ACT_NONE);
        k_gru<<<rows, H, 0, stream>>>(gib, ghb, hstate, H);
    };

    // ===== phase 0: xh = lrelu(x @ lin1^T + b) ==========================
    gemm(x, Fn, nullptr, 0, Fn, nullptr, lin1_w, Fn, lin1_b, xh, H, 0, N, H, Fn,
         AM_PLAIN, ACT_LRELU);

    // ===== phase 1: GATEConv + GRU0 =====================================
    k_rowdot<<<(N + 7) / 8, 256, 0, stream>>>(xh, H, gate_att_r, ns1, N, H, nullptr);
    // m = lrelu(concat(xh[src], edge_attr) @ gate_lin1^T)     [E x 256]
    gemm(xh, H, edge_attr, Fe, H, src, gate_lin1w, H + Fe, nullptr,
         eA, H, 0, E, H, H + Fe, AM_PLAIN, ACT_LRELU);
    // logits a = lrelu(m.gate_att_l + r[dst])
    k_rowdot<<<(E + 7) / 8, 256, 0, stream>>>(eA, H, gate_att_l, aE, E, H, nullptr);
    k_logit<<<(E + 255) / 256, 256, 0, stream>>>(aE, E, aE, ns1, dst, nullptr, nullptr);
    segsoftmax(aE, wE, dst, E, N);
    // t = m @ gate_lin2^T ;  hagg = gate_bias + scatter(t * w, dst)
    gemm(eA, H, nullptr, 0, H, nullptr, gate_lin2w, H, nullptr, eB, H, 0, E, H, H,
         AM_PLAIN, ACT_NONE);
    k_bias_rows<<<N, H, 0, stream>>>(hagg, gate_bias, H);
    k_scatter<<<E, H, 0, stream>>>(eB, nullptr, dst, wE, hagg, H);
    gru(hagg, xh, N, 0, gi, gh);

    // ===== phase 2: two GATConv layers + GRUs ===========================
    for (int l = 0; l < 2; ++l) {
        const float* awl = atom_w + (size_t)l * H * H;
        gemm(xh, H, nullptr, 0, H, nullptr, awl, H, nullptr, eA /*xl*/, H, 0,
             N, H, H, AM_PLAIN, ACT_NONE);
        k_rowdot<<<(N + 7) / 8, 256, 0, stream>>>(eA, H, atom_asrc + l * H, ns1, N, H, nullptr);
        k_rowdot<<<(N + 7) / 8, 256, 0, stream>>>(eA, H, atom_adst + l * H, ns2, N, H, nullptr);
        k_logit<<<(E + 255) / 256, 256, 0, stream>>>(aE, E, nullptr, ns1, src, ns2, dst);
        segsoftmax(aE, wE, dst, E, N);
        k_bias_rows<<<N, H, 0, stream>>>(hagg, atom_bias + l * H, H);
        k_scatter<<<E, H, 0, stream>>>(eA, src, dst, wE, hagg, H);
        gru(hagg, xh, N, 1 + l, gi, gh);
    }

    // ===== phase 3: attentive readout (3 timesteps) =====================
    k_fill<<<((int)((size_t)G * H + 255) / 256), 256, 0, stream>>>(outG, 0.f, G * H);
    k_scatter<<<N, H, 0, stream>>>(xh, nullptr, batch, nullptr, outG, H);
    k_relu<<<((int)((size_t)G * H + 255) / 256), 256, 0, stream>>>(outG, (long long)G * H);
    // xm = xh @ mol_w^T (fixed across timesteps), a_src[n] = xm . mol_att_src
    float* xm = big;   // edge/GRU buffers dead here; N*H fits
    gemm(xh, H, nullptr, 0, H, nullptr, mol_w, H, nullptr, xm, H, 0, N, H, H,
         AM_PLAIN, ACT_NONE);
    k_rowdot<<<(N + 7) / 8, 256, 0, stream>>>(xm, H, mol_asrc, ns1, N, H, nullptr);
    for (int t = 0; t < 3; ++t) {
        gemm(outG, H, nullptr, 0, H, nullptr, mol_w, H, nullptr, tmpG, H, 0, G, H, H,
             AM_PLAIN, ACT_NONE);
        k_rowdot<<<(G + 7) / 8, 256, 0, stream>>>(tmpG, H, mol_adst, dG, G, H, nullptr);
        k_logit<<<(N + 255) / 256, 256, 0, stream>>>(ns2, N, ns1, dG, batch, nullptr, nullptr);
        segsoftmax(ns2, aE /*w, E>=N*/, batch, N, G);
        k_bias_rows<<<G, H, 0, stream>>>(tmpG, mol_bias, H);
        k_scatter<<<N, H, 0, stream>>>(xm, nullptr, batch, aE, tmpG, H);
        gru(tmpG, outG, G, 3, giG, ghG);
    }

    // ===== phase 4: heads ==============================================
    gemm(outG, H, nullptr, 0, H, nullptr, lin2_w, H, lin2_b, fused, 512, 0,
         G, H, H, AM_PLAIN, ACT_NONE);
    gemm(mol_feats, Fm, nullptr, 0, Fm, nullptr, fcm_w[0], Fm, fcm_b[0], mA, H, 0,
         G, H, Fm, AM_PLAIN, ACT_SILU);
    gemm(mA, H, nullptr, 0, H, nullptr, fcm_w[1], H, fcm_b[1], mB, H, 0,
         G, H, H, AM_PLAIN, ACT_SILU);
    gemm(mB, H, nullptr, 0, H, nullptr, fcm_w[2], H, fcm_b[2], fused, 512, 256,
         G, H, H, AM_PLAIN, ACT_SILU);
    gemm(fused, 512, nullptr, 0, 512, nullptr, fc0_w, 512, fc0_b, mA, 512, 0,
         G, 512, 512, AM_PLAIN, ACT_SILU);
    gemm(mA, 512, nullptr, 0, 512, nullptr, fc1_w, 512, fc1_b, mB, 512, 0,
         G, 512, 512, AM_PLAIN, ACT_SILU);
    gemm(mB, 512, nullptr, 0, 512, nullptr, fc2_w, 512, fc2_b, mA, 512, 0,
         G, 512, 512, AM_PLAIN, ACT_SILU);
    gemm(mA, 512, nullptr, 0, 512, nullptr, fc3_w, 512, fc3_b, h128, 128, 0,
         G, 128, 512, AM_PLAIN, ACT_SILU);
    k_rowdot<<<(G + 7) / 8, 256, 0, stream>>>(h128, 128, fc4_w, out, G, 128, fc4_b);

    (void)n_in; (void)out_size; (void)ws_size; (void)dG;
}
